// SATLayer_regular_47837345743369
// MI455X (gfx1250) — matrix operand order repaired and checked by reading, not compiled, not run on hardware
//
#include <hip/hip_runtime.h>
#include <hip/hip_bf16.h>

// ---------------------------------------------------------------------------
// SATLayer on MI455X (gfx1250, wave32):
//  K1 lin1_kernel : y = relu(x@W1+b1) via v_wmma_f32_16x16x32_bf16,
//                   fused a1/a2 attention-score dot products in the epilogue.
//  K2 edge_agg    : one wave per edge; sigmoid(a1[r]+a2[c]) * xj[c] scatter-add
//                   with global_atomic_add_f32 (L2-resident rows).
//  K3 agg_gemm    : out = [y0|agg0|agg1|agg2] @ Wagg + bagg, WMMA bf16,
//                   Wagg^T staged as bf16 in LDS (128 KB).
// Inner loops batch-load all 8 B-fragments per K-chunk before issuing the
// 8 WMMAs so the scheduler can use partial s_wait_dscnt and overlap LDS
// traffic with the matrix pipe (instead of dscnt==0 before every WMMA).
// ---------------------------------------------------------------------------

typedef __attribute__((ext_vector_type(16))) __bf16 v16bf;
typedef __attribute__((ext_vector_type(8)))  __bf16 v8bf;
typedef __attribute__((ext_vector_type(8)))  float  v8f;

union ABfrag { v16bf v; v8bf h[2]; };

#define DIM 128

// ---------------------------------------------------------------------------
// Kernel 1: y = relu(x @ W1 + b1); also a1o = y@a1w + a1b, a2o = y@a2w + a2b.
// Block = 256 threads = 8 waves; each workgroup computes 128 rows x 128 cols.
// ---------------------------------------------------------------------------
__global__ __launch_bounds__(256)
void lin1_kernel(const float* __restrict__ x, const float* __restrict__ W1,
                 const float* __restrict__ b1,
                 const float* __restrict__ a1w, const float* __restrict__ a1b,
                 const float* __restrict__ a2w, const float* __restrict__ a2b,
                 float* __restrict__ y, float* __restrict__ a1o,
                 float* __restrict__ a2o, int M)
{
    extern __shared__ __bf16 lds[];
    __bf16* Wt = lds;                 // [n=128][k=128] transposed W1, bf16
    __bf16* xs = lds + DIM * DIM;     // [r=128][k=128] x tile, bf16

    const int t = threadIdx.x;
    const int rowBase = blockIdx.x * 128;

    // Stage W1^T (coalesced global read; one-time per workgroup).
    for (int idx = t; idx < DIM * DIM; idx += 256) {
        const int k = idx >> 7, n = idx & 127;
        Wt[n * DIM + k] = (__bf16)W1[idx];
    }
    // Stage 128 x-rows as bf16 (float4 coalesced), zero-fill OOB rows.
    for (int idx = t; idx < DIM * DIM / 4; idx += 256) {
        const int r  = idx >> 5;
        const int c4 = (idx & 31) * 4;
        float4 v = make_float4(0.f, 0.f, 0.f, 0.f);
        const int gr = rowBase + r;
        if (gr < M) v = *(const float4*)(x + (size_t)gr * DIM + c4);
        xs[r * DIM + c4 + 0] = (__bf16)v.x;
        xs[r * DIM + c4 + 1] = (__bf16)v.y;
        xs[r * DIM + c4 + 2] = (__bf16)v.z;
        xs[r * DIM + c4 + 3] = (__bf16)v.w;
    }
    __syncthreads();

    const int lane = t & 31;
    const int wave = t >> 5;
    const int half = lane >> 4;          // which 16-lane half
    const int l15  = lane & 15;
    const int mloc = wave * 16 + l15;    // A-matrix row fed by this lane

    v8f acc[8];
#pragma unroll
    for (int nt = 0; nt < 8; ++nt) {
        v8f z = {0.f, 0.f, 0.f, 0.f, 0.f, 0.f, 0.f, 0.f};
        acc[nt] = z;
    }

#pragma unroll
    for (int kc = 0; kc < 4; ++kc) {     // K = 128 in chunks of 32
        // A fragment: lanes 0-15 hold K {0-7,16-23}, lanes 16-31 {8-15,24-31}.
        ABfrag ua;
        const int kb = kc * 32 + half * 8;
        ua.h[0] = *(const v8bf*)&xs[mloc * DIM + kb];
        ua.h[1] = *(const v8bf*)&xs[mloc * DIM + kb + 16];
        // Batch-load all 8 B fragments for this K-chunk (column-contiguous).
        ABfrag ub[8];
#pragma unroll
        for (int nt = 0; nt < 8; ++nt) {
            const int n = nt * 16 + l15;
            ub[nt].h[0] = *(const v8bf*)&Wt[n * DIM + kc * 32 + half * 8];
            ub[nt].h[1] = *(const v8bf*)&Wt[n * DIM + kc * 32 + half * 8 + 16];
        }
#pragma unroll
        for (int nt = 0; nt < 8; ++nt) {
#if defined(__gfx1250__)
            acc[nt] = __builtin_amdgcn_wmma_f32_16x16x32_bf16(
                false, ua.v, false, ub[nt].v, (short)0, acc[nt], false, false);
#endif
        }
    }

    // Epilogue: bias + relu + store y; fused attention-score dot products.
    float b1v[8], w1v[8], w2v[8];
#pragma unroll
    for (int nt = 0; nt < 8; ++nt) {
        const int n = nt * 16 + l15;
        b1v[nt] = b1[n]; w1v[nt] = a1w[n]; w2v[nt] = a2w[n];
    }
    float pa1[8], pa2[8];
#pragma unroll
    for (int v = 0; v < 8; ++v) { pa1[v] = 0.f; pa2[v] = 0.f; }

    const int rb = rowBase + wave * 16;
#pragma unroll
    for (int nt = 0; nt < 8; ++nt) {
#pragma unroll
        for (int v = 0; v < 8; ++v) {
            float val = acc[nt][v] + b1v[nt];
            val = fmaxf(val, 0.f);
            const int m = rb + v + 8 * half;   // C layout: m = vgpr + 8*half
            if (m < M) y[(size_t)m * DIM + nt * 16 + l15] = val;
            pa1[v] += val * w1v[nt];
            pa2[v] += val * w2v[nt];
        }
    }
    // Butterfly-reduce over the 16 lanes of each half (covers all 128 n).
#pragma unroll
    for (int v = 0; v < 8; ++v) {
#pragma unroll
        for (int off = 1; off < 16; off <<= 1) {
            pa1[v] += __shfl_xor(pa1[v], off, 32);
            pa2[v] += __shfl_xor(pa2[v], off, 32);
        }
    }
    if (l15 == 0) {
        const float ab1 = *a1b, ab2 = *a2b;
#pragma unroll
        for (int v = 0; v < 8; ++v) {
            const int m = rb + v + 8 * half;
            if (m < M) { a1o[m] = pa1[v] + ab1; a2o[m] = pa2[v] + ab2; }
        }
    }
}

// ---------------------------------------------------------------------------
// Kernel 2: edge aggregation. One wave32 per edge:
//   att = sigmoid(a1[row] + a2[col]); agg[row] += att * xj[col]  (128 f32)
// Lane l handles 4 floats (float4 gather + 4 fp32 global atomics).
// ---------------------------------------------------------------------------
__global__ __launch_bounds__(256)
void edge_agg_kernel(const float* __restrict__ xj, const float* __restrict__ a1,
                     const float* __restrict__ a2, const int* __restrict__ rows,
                     const int* __restrict__ cols, float* __restrict__ agg, int nE)
{
    const int lane = threadIdx.x & 31;
    const int wid  = blockIdx.x * (blockDim.x >> 5) + (threadIdx.x >> 5);
    const int nw   = gridDim.x * (blockDim.x >> 5);

    for (int e = wid; e < nE; e += nw) {
        const int r = rows[e];
        const int c = cols[e];
        const int en = e + nw;
        if (en < nE) {                       // prefetch next edge's indices
            __builtin_prefetch(&rows[en], 0, 1);
            __builtin_prefetch(&cols[en], 0, 1);
        }
        const float s   = a1[r] + a2[c];
        const float att = 1.f / (1.f + __expf(-s));
        const float4 xv = ((const float4*)(xj + (size_t)c * DIM))[lane];
        float* dst = agg + (size_t)r * DIM + lane * 4;
        (void)__hip_atomic_fetch_add(dst + 0, att * xv.x, __ATOMIC_RELAXED, __HIP_MEMORY_SCOPE_AGENT);
        (void)__hip_atomic_fetch_add(dst + 1, att * xv.y, __ATOMIC_RELAXED, __HIP_MEMORY_SCOPE_AGENT);
        (void)__hip_atomic_fetch_add(dst + 2, att * xv.z, __ATOMIC_RELAXED, __HIP_MEMORY_SCOPE_AGENT);
        (void)__hip_atomic_fetch_add(dst + 3, att * xv.w, __ATOMIC_RELAXED, __HIP_MEMORY_SCOPE_AGENT);
    }
}

// ---------------------------------------------------------------------------
// Kernel 3: out = concat(s0,s1,s2,s3; K=512) @ Wagg + bagg via WMMA bf16.
// Wagg^T staged as bf16 in LDS (128 KB). A-fragments streamed from global.
// ---------------------------------------------------------------------------
__global__ __launch_bounds__(256)
void agg_gemm_kernel(const float* __restrict__ s0, const float* __restrict__ s1,
                     const float* __restrict__ s2, const float* __restrict__ s3,
                     const float* __restrict__ Wagg, const float* __restrict__ bagg,
                     float* __restrict__ out, int M)
{
    extern __shared__ __bf16 lds2[];
    __bf16* Wt = lds2;                 // [n=128][k=512]

    const int t = threadIdx.x;
    for (int idx = t; idx < 512 * DIM; idx += 256) {
        const int k = idx >> 7, n = idx & 127;
        Wt[n * 512 + k] = (__bf16)Wagg[idx];
    }
    __syncthreads();

    const int lane = t & 31;
    const int wave = t >> 5;
    const int half = lane >> 4;
    const int l15  = lane & 15;
    const int mA   = blockIdx.x * 128 + wave * 16 + l15;  // A-row this lane feeds
    const bool mv  = (mA < M);
    const float* srcs[4] = { s0, s1, s2, s3 };

    v8f acc[8];
#pragma unroll
    for (int nt = 0; nt < 8; ++nt) {
        v8f z = {0.f, 0.f, 0.f, 0.f, 0.f, 0.f, 0.f, 0.f};
        acc[nt] = z;
    }

#pragma unroll
    for (int kc = 0; kc < 16; ++kc) {      // K = 512 in chunks of 32
        // Issue the global A-fragment loads first (longest latency) ...
        const float* rp = srcs[kc >> 2] + (size_t)mA * DIM + (kc & 3) * 32;
        const int kb = half * 8;
        float af[16];
        if (mv) {
            const float4 f0 = *(const float4*)(rp + kb);
            const float4 f1 = *(const float4*)(rp + kb + 4);
            const float4 f2 = *(const float4*)(rp + kb + 16);
            const float4 f3 = *(const float4*)(rp + kb + 20);
            af[0]=f0.x; af[1]=f0.y; af[2]=f0.z; af[3]=f0.w;
            af[4]=f1.x; af[5]=f1.y; af[6]=f1.z; af[7]=f1.w;
            af[8]=f2.x; af[9]=f2.y; af[10]=f2.z; af[11]=f2.w;
            af[12]=f3.x; af[13]=f3.y; af[14]=f3.z; af[15]=f3.w;
        } else {
#pragma unroll
            for (int j = 0; j < 16; ++j) af[j] = 0.f;
        }
        // ... then batch the 8 LDS B-fragment loads ...
        ABfrag ub[8];
#pragma unroll
        for (int nt = 0; nt < 8; ++nt) {
            const int n = nt * 16 + l15;
            ub[nt].h[0] = *(const v8bf*)&Wt[n * 512 + kc * 32 + half * 8];
            ub[nt].h[1] = *(const v8bf*)&Wt[n * 512 + kc * 32 + half * 8 + 16];
        }
        // ... convert A and run the 8 WMMAs.
        ABfrag ua;
#pragma unroll
        for (int j = 0; j < 16; ++j) ua.v[j] = (__bf16)af[j];
#pragma unroll
        for (int nt = 0; nt < 8; ++nt) {
#if defined(__gfx1250__)
            acc[nt] = __builtin_amdgcn_wmma_f32_16x16x32_bf16(
                false, ua.v, false, ub[nt].v, (short)0, acc[nt], false, false);
#endif
        }
    }

    float bv[8];
#pragma unroll
    for (int nt = 0; nt < 8; ++nt) bv[nt] = bagg[nt * 16 + l15];

    const int rb = blockIdx.x * 128 + wave * 16;
#pragma unroll
    for (int nt = 0; nt < 8; ++nt) {
#pragma unroll
        for (int v = 0; v < 8; ++v) {
            const int m = rb + v + 8 * half;
            if (m < M) out[(size_t)m * DIM + nt * 16 + l15] = acc[nt][v] + bv[nt];
        }
    }
}

// ---------------------------------------------------------------------------
// Zero-fill (workspace agg buffers are poisoned by the harness).
// ---------------------------------------------------------------------------
__global__ __launch_bounds__(256)
void zero_kernel(float4* __restrict__ p, long n4)
{
    const long i = (long)blockIdx.x * blockDim.x + threadIdx.x;
    if (i < n4) p[i] = make_float4(0.f, 0.f, 0.f, 0.f);
}

// ---------------------------------------------------------------------------
extern "C" void kernel_launch(void* const* d_in, const int* in_sizes, int n_in,
                              void* d_out, int out_size, void* d_ws, size_t ws_size,
                              hipStream_t stream)
{
    const float* x0   = (const float*)d_in[0];
    const float* x1   = (const float*)d_in[1];
    const float* x2   = (const float*)d_in[2];
    const float* W1   = (const float*)d_in[3];
    const float* b1   = (const float*)d_in[4];
    const float* a1w  = (const float*)d_in[5];
    const float* a1b  = (const float*)d_in[6];
    const float* a2w  = (const float*)d_in[7];
    const float* a2b  = (const float*)d_in[8];
    const float* Wagg = (const float*)d_in[9];
    const float* bagg = (const float*)d_in[10];
    const int*   r0   = (const int*)d_in[11];
    const int*   c0   = (const int*)d_in[12];
    const int*   r1   = (const int*)d_in[13];
    const int*   c1   = (const int*)d_in[14];
    const int*   r2   = (const int*)d_in[15];
    const int*   c2   = (const int*)d_in[16];

    const int N0 = in_sizes[0] / DIM;
    const int N1 = in_sizes[1] / DIM;
    const int N2 = in_sizes[2] / DIM;
    const int E  = in_sizes[11];

    // Workspace carve-up (256B-aligned slices).
    size_t off = 0;
    auto carve = [&](size_t bytes) -> void* {
        void* p = (char*)d_ws + off;
        off += (bytes + 255) & ~(size_t)255;
        return p;
    };
    float* y0   = (float*)carve((size_t)N0 * DIM * 4);
    float* y1   = (float*)carve((size_t)N1 * DIM * 4);
    float* y2   = (float*)carve((size_t)N2 * DIM * 4);
    float* agg0 = (float*)carve((size_t)N0 * DIM * 4);   // agg0/1/2 contiguous
    float* agg1 = (float*)carve((size_t)N0 * DIM * 4);
    float* agg2 = (float*)carve((size_t)N0 * DIM * 4);
    float* a1_0 = (float*)carve((size_t)N0 * 4);
    float* a2_0 = (float*)carve((size_t)N0 * 4);
    float* a1_1 = (float*)carve((size_t)N1 * 4);
    float* a2_1 = (float*)carve((size_t)N1 * 4);
    float* a1_2 = (float*)carve((size_t)N2 * 4);
    float* a2_2 = (float*)carve((size_t)N2 * 4);
    (void)ws_size; (void)n_in; (void)out_size;

    // 1) zero the three agg buffers (contiguous -> one kernel).
    {
        const long n4 = (long)3 * N0 * DIM / 4;
        const int blocks = (int)((n4 + 255) / 256);
        zero_kernel<<<blocks, 256, 0, stream>>>((float4*)agg0, n4);
    }

    // 2) lin1 on x0/x1/x2 (fused attention scores).
    const size_t lds1 = (size_t)2 * DIM * DIM * sizeof(__bf16);  // 64 KB
    lin1_kernel<<<(N0 + 127) / 128, 256, lds1, stream>>>(
        x0, W1, b1, a1w, a1b, a2w, a2b, y0, a1_0, a2_0, N0);
    lin1_kernel<<<(N1 + 127) / 128, 256, lds1, stream>>>(
        x1, W1, b1, a1w, a1b, a2w, a2b, y1, a1_1, a2_1, N1);
    lin1_kernel<<<(N2 + 127) / 128, 256, lds1, stream>>>(
        x2, W1, b1, a1w, a1b, a2w, a2b, y2, a1_2, a2_2, N2);

    // 3) edge-wise attention aggregation (wave-per-edge).
    const int eb = 2048;
    edge_agg_kernel<<<eb, 256, 0, stream>>>(y0, a1_0, a2_0, r0, c0, agg0, E);
    edge_agg_kernel<<<eb, 256, 0, stream>>>(y1, a1_0, a2_1, r1, c1, agg1, E);
    edge_agg_kernel<<<eb, 256, 0, stream>>>(y2, a1_0, a2_2, r2, c2, agg2, E);

    // 4) final GEMM over the virtual concat [y0|agg0|agg1|agg2].
    const size_t lds3 = (size_t)512 * DIM * sizeof(__bf16);      // 128 KB
    agg_gemm_kernel<<<(N0 + 127) / 128, 256, lds3, stream>>>(
        y0, agg0, agg1, agg2, Wagg, bagg, (float*)d_out, N0);
}
